// MultiHeadNetwork_33595234189689
// MI455X (gfx1250) — matrix operand order repaired and checked by reading, not compiled, not run on hardware
//
#include <hip/hip_runtime.h>

typedef __attribute__((ext_vector_type(16))) _Float16 v16h;
typedef __attribute__((ext_vector_type(8)))  float    v8f;
typedef unsigned int u32;

union FragU { uint4 q[2]; v16h h; };

// ---------------------------------------------------------------------------
// Pack a row-major f32 weight matrix W[K][512] into WMMA-B-fragment order,
// f16, K padded to Kpad (multiple of 32). Word layout (32-bit = 2 packed f16,
// low = even k, high = odd k):
//   id = (((ntile*G + g)*2 + half)*16 + nlane)*8 + v
//   k  = g*32 + half*16 + 2*v , n = ntile*16 + nlane , G = Kpad/32
// so each lane's B fragment for one 16x16x32 WMMA is 8 consecutive dwords.
// ---------------------------------------------------------------------------
__global__ void pack_weight_kernel(const float* __restrict__ W, u32* __restrict__ out,
                                   int Kin, int Kpad) {
    const int G = Kpad >> 5;
    const int total = (Kpad >> 1) * 512;
    for (int id = blockIdx.x * blockDim.x + threadIdx.x; id < total;
         id += gridDim.x * blockDim.x) {
        int v  = id & 7;
        int nl = (id >> 3) & 15;
        int hh = (id >> 7) & 1;
        int gt = id >> 8;           // = t*G + g
        int g  = gt % G;
        int t  = gt / G;
        int k  = g * 32 + hh * 16 + 2 * v;
        int n  = t * 16 + nl;
        float lo = (k     < Kin) ? W[(size_t)k       * 512 + n] : 0.0f;
        float hi = (k + 1 < Kin) ? W[(size_t)(k + 1) * 512 + n] : 0.0f;
        _Float16 l16 = (_Float16)lo, h16 = (_Float16)hi;
        u32 w = ((u32)__builtin_bit_cast(unsigned short, h16) << 16)
              |  (u32)__builtin_bit_cast(unsigned short, l16);
        out[id] = w;
    }
}

// head_W (50, 512, 8) f32  ->  headWt[t][j][k] f16 (contiguous over k)
__global__ void pack_head_kernel(const float* __restrict__ hw, _Float16* __restrict__ out) {
    int id = blockIdx.x * blockDim.x + threadIdx.x;   // exact grid: 50*8*512
    int t = id >> 12;
    int j = (id >> 9) & 7;
    int k = id & 511;
    out[id] = (_Float16)hw[(size_t)t * 4096 + (size_t)k * 8 + j];
}

// ---------------------------------------------------------------------------
// One dense layer: out[64][512] = act( in[64][KS] @ W[KS][512] + b (+ resid) )
// 8 waves: wave w -> (wy = w>>2) covers M-tiles {2wy, 2wy+1},
//                    (wx = w&3)  covers N-tiles {8wx .. 8wx+7}.
// A fragment (contiguous k order, as packed B): lane m = lane&15, half = lane>>4,
//   dwords v0..7 at k = g*32 + half*16 + {0,2,..,14}.
//   => two contiguous 16B LDS loads at word offsets (m*KS/2 + g*16 + half*8)
//      and +4.
// ---------------------------------------------------------------------------
template <int KS, int G, bool RESID>
__device__ __forceinline__ void mlp_layer(const _Float16* __restrict__ in,
                                          const u32* __restrict__ Wp,
                                          const float* __restrict__ bias,
                                          _Float16* __restrict__ outb) {
    const int tid  = threadIdx.x;
    const int lane = tid & 31;
    const int wv   = tid >> 5;
    const int wy   = wv >> 2;   // 0..1
    const int wx   = wv & 3;    // 0..3
    const int half = lane >> 4;
    const int nl   = lane & 15;

    v8f acc[2][8] = {};

    const u32* ldsW = (const u32*)in;

    for (int g = 0; g < G; ++g) {
        // A fragments for this wave's two M-tiles
        v16h afr[2];
#pragma unroll
        for (int i = 0; i < 2; ++i) {
            const int mrow = (wy * 2 + i) * 16 + nl;
            const u32* p = ldsW + mrow * (KS >> 1) + g * 16 + half * 8;
            FragU f;
            f.q[0] = *(const uint4*)(p);
            f.q[1] = *(const uint4*)(p + 4);
            afr[i] = f.h;
        }
        if (g + 1 < G) {
            const u32* pf = Wp + ((((size_t)(wx * 8) * G + (g + 1)) * 2 + half) * 16 + nl) * 8;
            __builtin_prefetch(pf, 0, 0);   // global_prefetch_b8 for next K-step
        }
#pragma unroll
        for (int j = 0; j < 8; ++j) {
            const int nt = wx * 8 + j;
            const u32* q = Wp + ((((size_t)nt * G + g) * 2 + half) * 16 + nl) * 8;
            FragU fb;
            fb.q[0] = *(const uint4*)(q);
            fb.q[1] = *(const uint4*)(q + 4);
            acc[0][j] = __builtin_amdgcn_wmma_f32_16x16x32_f16(
                false, afr[0], false, fb.h, (short)0, acc[0][j], false, false);
            acc[1][j] = __builtin_amdgcn_wmma_f32_16x16x32_f16(
                false, afr[1], false, fb.h, (short)0, acc[1][j], false, false);
        }
    }

    // Epilogue: bias (+ residual) + ReLU, store f16 to LDS out (stride 512).
    // C layout: VGPR r -> lanes 0-15: M=r, N=lane ; lanes 16-31: M=r+8.
#pragma unroll
    for (int i = 0; i < 2; ++i) {
#pragma unroll
        for (int j = 0; j < 8; ++j) {
            const int n  = (wx * 8 + j) * 16 + nl;
            const float bv = bias[n];
            const int mbase = (wy * 2 + i) * 16 + half * 8;
#pragma unroll
            for (int r = 0; r < 8; ++r) {
                const int mrow = mbase + r;
                float v = acc[i][j][r] + bv;
                if (RESID) v += (float)in[mrow * 512 + n];
                v = fmaxf(v, 0.0f);
                outb[mrow * 512 + n] = (_Float16)v;
            }
        }
    }
}

// ---------------------------------------------------------------------------
// Fused network: 64 rows per block; h stays in LDS across all 3 layers;
// only the argmax-selected head is computed.
// ---------------------------------------------------------------------------
__global__ __launch_bounds__(256) void fused_mlp_kernel(
    const float* __restrict__ x,
    const float* __restrict__ b0, const float* __restrict__ b1,
    const float* __restrict__ b2,
    const u32* __restrict__ W0p, const u32* __restrict__ W1p,
    const u32* __restrict__ W2p,
    const _Float16* __restrict__ headWt, const float* __restrict__ head_b,
    float* __restrict__ out) {
    extern __shared__ char smem_raw[];
    _Float16* bufA = (_Float16*)smem_raw;       // 64*512 f16 = 64 KB
    _Float16* bufB = bufA + 64 * 512;           // 64 KB
    int* taskLds   = (int*)(bufB + 64 * 512);   // 64 ints

    const int tid  = threadIdx.x;
    const int row0 = blockIdx.x * 64;

    // Stage x tile -> bufA as f16, K padded 114 -> 128 (zeros), row stride 128.
    for (int i = tid; i < 64 * 128; i += 256) {
        int m = i >> 7, k = i & 127;
        float v = (k < 114) ? x[(size_t)(row0 + m) * 114 + k] : 0.0f;
        bufA[m * 128 + k] = (_Float16)v;
    }
    // Per-row task = argmax over x[:, 64..113] (first max wins, as jnp.argmax).
    if (tid < 64) {
        const float* xr = x + (size_t)(row0 + tid) * 114 + 64;
        int best = 0;
        float bv = xr[0];
        for (int c = 1; c < 50; ++c) {
            float v = xr[c];
            if (v > bv) { bv = v; best = c; }
        }
        taskLds[tid] = best;
    }
    __syncthreads();

    mlp_layer<128, 4, false>(bufA, W0p, b0, bufB);   // layer 0
    __syncthreads();
    mlp_layer<512, 16, true>(bufB, W1p, b1, bufA);   // layer 1 (+resid)
    __syncthreads();
    mlp_layer<512, 16, true>(bufA, W2p, b2, bufB);   // layer 2 (+resid)
    __syncthreads();

    // Selected head only: each thread computes 2 of the 8 outputs of one row.
    const int m  = tid >> 2;
    const int j0 = (tid & 3) * 2;
    const int t  = taskLds[m];
    const _Float16* hrow = bufB + m * 512;
    const _Float16* w0p  = headWt + ((size_t)t * 8 + j0) * 512;
    const _Float16* w1p  = w0p + 512;
    float a0 = 0.0f, a1 = 0.0f;
    for (int k = 0; k < 512; ++k) {
        float hv = (float)hrow[k];
        a0 += hv * (float)w0p[k];
        a1 += hv * (float)w1p[k];
    }
    size_t ob = ((size_t)(row0 + m)) * 8 + j0;
    out[ob]     = a0 + head_b[t * 8 + j0];
    out[ob + 1] = a1 + head_b[t * 8 + j0 + 1];
}

extern "C" void kernel_launch(void* const* d_in, const int* in_sizes, int n_in,
                              void* d_out, int out_size, void* d_ws, size_t ws_size,
                              hipStream_t stream) {
    (void)in_sizes; (void)n_in; (void)out_size; (void)ws_size;
    const float* x      = (const float*)d_in[0];
    const float* W0     = (const float*)d_in[1];
    const float* b0     = (const float*)d_in[2];
    const float* W1     = (const float*)d_in[3];
    const float* b1     = (const float*)d_in[4];
    const float* W2     = (const float*)d_in[5];
    const float* b2     = (const float*)d_in[6];
    const float* headW  = (const float*)d_in[7];
    const float* head_b = (const float*)d_in[8];

    char* ws = (char*)d_ws;
    u32* W0p = (u32*)ws;                                   // 64*512  dwords = 128 KB
    u32* W1p = (u32*)(ws + (128 << 10));                   // 256*512 dwords = 512 KB
    u32* W2p = (u32*)(ws + (128 << 10) + (512 << 10));     // 512 KB
    _Float16* headWt = (_Float16*)(ws + (128 << 10) + 2 * (512 << 10));  // 400 KB

    pack_weight_kernel<<<128, 256, 0, stream>>>(W0, W0p, 114, 128);
    pack_weight_kernel<<<512, 256, 0, stream>>>(W1, W1p, 512, 512);
    pack_weight_kernel<<<512, 256, 0, stream>>>(W2, W2p, 512, 512);
    pack_head_kernel<<<800, 256, 0, stream>>>(headW, headWt);

    const size_t shmem = 2 * 64 * 512 * sizeof(_Float16) + 64 * sizeof(int);
    fused_mlp_kernel<<<1024, 256, shmem, stream>>>(
        x, b0, b1, b2, W0p, W1p, W2p, headWt, head_b, (float*)d_out);
}
